// TSTEncoder_11682311045243
// MI455X (gfx1250) — matrix operand order repaired and checked by reading, not compiled, not run on hardware
//
#include <hip/hip_runtime.h>
#include <hip/hip_bf16.h>
#include <math.h>
#include <stdint.h>

// Problem constants (match reference)
#define LYR 4
#define DIM 512
#define NH  8
#define FFD 2048
#define BB  32
#define SS  512
#define DKH 64            // DIM / NH
#define MTOK (BB * SS)    // 16384 token rows
#define BH  (BB * NH)     // 256 (batch*heads)
#define EPSV 1e-5f

typedef __attribute__((ext_vector_type(16))) __bf16 bf16x16;
typedef __attribute__((ext_vector_type(8)))  __bf16 bf16x8;
typedef __attribute__((ext_vector_type(4)))  __bf16 bf16x4;
typedef __attribute__((ext_vector_type(8)))  float  f32x8;
typedef __attribute__((ext_vector_type(4)))  float  f32x4;

#define FLAG_GELU 1

// ---------------------------------------------------------------------------
// fp32 -> bf16 conversion (grid-stride)
// ---------------------------------------------------------------------------
__global__ void f32_to_bf16_kernel(const float* __restrict__ in,
                                   __bf16* __restrict__ out, int n) {
  int i = blockIdx.x * blockDim.x + threadIdx.x;
  int stride = gridDim.x * blockDim.x;
  for (; i < n; i += stride) out[i] = (__bf16)in[i];
}

// ---------------------------------------------------------------------------
// fp32 [K,N] -> bf16 transposed [N,K], per-layer batch in z.
// 32x32 tile through LDS; coalesced 16B f32 loads, coalesced 8B bf16 stores.
// ---------------------------------------------------------------------------
__launch_bounds__(256)
__global__ void transpose_cvt_kernel(const float* __restrict__ in,
                                     __bf16* __restrict__ out, int K, int N) {
  __shared__ __bf16 tile[32][33];
  const size_t lofs = (size_t)blockIdx.z * K * N;
  const float* ip = in + lofs;
  __bf16* op = out + lofs;
  int n0 = blockIdx.x * 32;
  int k0 = blockIdx.y * 32;
  int t = threadIdx.x;
  int r = t >> 3;          // 0..31
  int c4 = (t & 7) * 4;    // 0..28
  f32x4 v = *(const f32x4*)&ip[(size_t)(k0 + r) * N + n0 + c4];
#pragma unroll
  for (int j = 0; j < 4; ++j) tile[r][c4 + j] = (__bf16)v[j];
  __syncthreads();
  bf16x4 o;
#pragma unroll
  for (int j = 0; j < 4; ++j) o[j] = tile[c4 + j][r];
  *(bf16x4*)&op[(size_t)(n0 + r) * K + k0 + c4] = o;
}

// ---------------------------------------------------------------------------
// V[b,kk,h,d] (bf16, token-major) -> Vt[bh, d, kk] so P*V B-fragments are
// K-contiguous. 32x32 tile through LDS, z = bh.
// ---------------------------------------------------------------------------
__launch_bounds__(256)
__global__ void transpose_v_kernel(const __bf16* __restrict__ V,
                                   __bf16* __restrict__ Vt) {
  __shared__ __bf16 tile[32][33];
  int bh = blockIdx.z;
  int b = bh >> 3, h = bh & (NH - 1);
  int kk0 = blockIdx.x * 32;
  int d0 = blockIdx.y * 32;
  int t = threadIdx.x;
  int r = t >> 3;          // 0..31
  int c4 = (t & 7) * 4;    // 0..28
  bf16x4 v = *(const bf16x4*)&V[(size_t)(b * SS + kk0 + r) * DIM + h * DKH + d0 + c4];
#pragma unroll
  for (int j = 0; j < 4; ++j) tile[r][c4 + j] = v[j];
  __syncthreads();
  bf16x4 o;
#pragma unroll
  for (int j = 0; j < 4; ++j) o[j] = tile[c4 + j][r];
  *(bf16x4*)&Vt[((size_t)bh * DKH + d0 + r) * SS + kk0 + c4] = o;
}

// ---------------------------------------------------------------------------
// Generic bf16 GEMM:  C[M,N] = A[M,K] * Bt[N,K]^T + bias, optional exact GELU.
// Bt is the TRANSPOSED operand ([N,K] row-major) so both A and B fragments are
// contiguous vector loads (CDNA5 ISA 7.12.2 16-bit layouts).
// Block tile 64(M) x 128(N), K-step 32. 256 threads = 8 waves (wave32).
// Staging uses the gfx1250 async Global->LDS path (GLOBAL_LOAD_ASYNC_TO_LDS_-
// B128, ASYNCcnt) with explicit double buffering: VGPR-free copies, stage s+1
// in flight while stage s is consumed by WMMA.
// ---------------------------------------------------------------------------
__launch_bounds__(256)
__global__ void gemm_bf16_kernel(const __bf16* __restrict__ A,
                                 const __bf16* __restrict__ Bt,
                                 const float* __restrict__ bias,
                                 float* __restrict__ Cf,
                                 __bf16* __restrict__ Cb,
                                 int M, int N, int K, int flags) {
  __shared__ __align__(64) __bf16 As[2][64][32];     // [buf][m][k]
  __shared__ __align__(64) __bf16 BsT[2][128][48];   // [buf][n][k], 96B rows

  const int tid  = threadIdx.x;
  const int wave = tid >> 5;
  const int lane = tid & 31;
  const int wm = wave & 3, wn = wave >> 2;
  const int row0 = blockIdx.y * 64;
  const int col0 = blockIdx.x * 128;
  const int half = lane >> 4;
  const int l15  = lane & 15;

  // staging coordinates
  const int ar = tid >> 2, ac = (tid & 3) * 8;    // A: 64 rows x 4 chunks of 8
  const int bn = tid >> 1, bk = (tid & 1) * 16;   // B: 128 rows x 2 chunks of 16

  // Per-thread async-stage: 3 x b128 Global->LDS (16B of A, 32B of B).
  // LDS offset = low 32 bits of the generic shared-pointer (LDS aperture).
  auto stage_async = [&](int kk, int buf) {
    unsigned lA = (unsigned)(uintptr_t)&As[buf][ar][ac];
    unsigned lB = (unsigned)(uintptr_t)&BsT[buf][bn][bk];
    unsigned long long ga =
        (unsigned long long)(uintptr_t)&A[(size_t)(row0 + ar) * K + kk + ac];
    unsigned long long gb =
        (unsigned long long)(uintptr_t)&Bt[(size_t)(col0 + bn) * K + kk + bk];
    asm volatile("global_load_async_to_lds_b128 %0, %1, off"
                 :: "v"(lA), "v"(ga) : "memory");
    asm volatile("global_load_async_to_lds_b128 %0, %1, off"
                 :: "v"(lB), "v"(gb) : "memory");
    asm volatile("global_load_async_to_lds_b128 %0, %1, off offset:16"
                 :: "v"(lB), "v"(gb) : "memory");
  };

  f32x8 acc[4] = {};

  const int nst = K / 32;
  stage_async(0, 0);
  for (int s = 0; s < nst; ++s) {
    const int cur = s & 1;
    const bool more = (s + 1 < nst);
    if (more) {
      stage_async((s + 1) * 32, cur ^ 1);  // prefetch next stage (in flight)
      // async ops complete in order: <=3 outstanding => stage s landed
      asm volatile("s_wait_asynccnt 0x3" ::: "memory");
    } else {
      asm volatile("s_wait_asynccnt 0x0" ::: "memory");
    }
    __syncthreads();  // all waves' stage-s data visible in LDS

    // A fragment: lane-half 0 holds K={0..15}, half 1 holds {16..31}
    bf16x16 afrag;
    {
      const __bf16* arow = &As[cur][wm * 16 + l15][0];
      afrag = *(const bf16x16*)&arow[half * 16];
    }
#pragma unroll
    for (int t = 0; t < 4; ++t) {
      // B fragment: lane = column n; K-contiguous -> single 32B LDS read
      int nl = wn * 64 + t * 16 + l15;
      bf16x16 bfrag = *(const bf16x16*)&BsT[cur][nl][half * 16];
      acc[t] = __builtin_amdgcn_wmma_f32_16x16x32_bf16(
          false, afrag, false, bfrag, (short)0, acc[t], false, false);
    }
    __syncthreads();  // done reading buf cur before it is re-staged
  }

  // Epilogue: D layout = lane column n, rows m = half*8 + v
#pragma unroll
  for (int t = 0; t < 4; ++t) {
    int c = col0 + wn * 64 + t * 16 + l15;
    float bv = bias ? bias[c] : 0.0f;
#pragma unroll
    for (int v = 0; v < 8; ++v) {
      int r = row0 + wm * 16 + half * 8 + v;
      float val = acc[t][v] + bv;
      if (flags & FLAG_GELU)
        val = 0.5f * val * (1.0f + erff(val * 0.70710678118654752f));
      size_t idx = (size_t)r * N + c;
      if (Cf) Cf[idx] = val;
      if (Cb) Cb[idx] = (__bf16)val;
    }
  }
}

// ---------------------------------------------------------------------------
// Attention scores: scores[b,h,q,k] (+)= scale * sum_d Q[b,q,h,d]*K[b,k,h,d]
// grid: x = S/16 q-tiles, y = S/128 k-chunks, z = B*H. 8 waves, one 16x16
// tile each; DK=64 -> two 16x16x32 WMMA steps. K rows contiguous in d, so the
// B fragment is a single 32-byte load.
// ---------------------------------------------------------------------------
__launch_bounds__(256)
__global__ void attn_scores_kernel(const __bf16* __restrict__ Q,
                                   const __bf16* __restrict__ Km,
                                   float* __restrict__ scores,
                                   int first, float scale) {
  int bh = blockIdx.z;
  int b = bh >> 3, h = bh & (NH - 1);
  int wave = threadIdx.x >> 5, lane = threadIdx.x & 31;
  int half = lane >> 4, l15 = lane & 15;
  int q0 = blockIdx.x * 16;
  int k0 = blockIdx.y * 128 + wave * 16;

  const __bf16* Qb = Q + ((size_t)b * SS) * DIM + h * DKH;
  const __bf16* Kb = Km + ((size_t)b * SS) * DIM + h * DKH;

  f32x8 acc = {};
#pragma unroll
  for (int ks = 0; ks < DKH; ks += 32) {
    bf16x16 afrag, bfrag;
    const __bf16* qrow = Qb + (size_t)(q0 + l15) * DIM + ks;
    afrag = *(const bf16x16*)&qrow[half * 16];
    const __bf16* krow = Kb + (size_t)(k0 + l15) * DIM + ks + half * 16;
    bfrag = *(const bf16x16*)krow;
    acc = __builtin_amdgcn_wmma_f32_16x16x32_bf16(
        false, afrag, false, bfrag, (short)0, acc, false, false);
  }

  size_t base = (size_t)bh * SS * SS;
  int c = k0 + l15;
#pragma unroll
  for (int v = 0; v < 8; ++v) {
    int r = q0 + half * 8 + v;
    size_t idx = base + (size_t)r * SS + c;
    float val = acc[v] * scale;
    if (!first) val += scores[idx];
    scores[idx] = val;  // residual pre-softmax scores persist across layers
  }
}

// ---------------------------------------------------------------------------
// Row softmax over S=512, fp32 in -> bf16 probs. One row per 256-thread block.
// ---------------------------------------------------------------------------
__launch_bounds__(256)
__global__ void softmax_kernel(const float* __restrict__ scores,
                               __bf16* __restrict__ probs) {
  __shared__ float red[256];
  size_t row = blockIdx.x;
  const float* sp = scores + row * SS;
  __bf16* pp = probs + row * SS;
  int t = threadIdx.x;
  float a = sp[t], b = sp[t + 256];
  red[t] = fmaxf(a, b);
  __syncthreads();
  for (int s = 128; s > 0; s >>= 1) {
    if (t < s) red[t] = fmaxf(red[t], red[t + s]);
    __syncthreads();
  }
  float mx = red[0];
  __syncthreads();
  float e0 = __expf(a - mx), e1 = __expf(b - mx);
  red[t] = e0 + e1;
  __syncthreads();
  for (int s = 128; s > 0; s >>= 1) {
    if (t < s) red[t] += red[t + s];
    __syncthreads();
  }
  float inv = 1.0f / red[0];
  pp[t]       = (__bf16)(e0 * inv);
  pp[t + 256] = (__bf16)(e1 * inv);
}

// ---------------------------------------------------------------------------
// Attention output: O[b,q,h,d] = sum_k P[b,h,q,k] * Vt[bh,d,k].
// With Vt K-contiguous, both fragments are direct global vector loads; no LDS.
// 256 threads = 8 waves, block covers 128 q rows x full DK=64.
// ---------------------------------------------------------------------------
__launch_bounds__(256)
__global__ void attn_out_kernel(const __bf16* __restrict__ P,
                                const __bf16* __restrict__ Vt,
                                __bf16* __restrict__ O) {
  int bh = blockIdx.y;
  int b = bh >> 3, h = bh & (NH - 1);
  int wave = threadIdx.x >> 5, lane = threadIdx.x & 31;
  int half = lane >> 4, l15 = lane & 15;
  int q0 = blockIdx.x * 128 + wave * 16;
  const __bf16* Pb  = P + (size_t)bh * SS * SS;
  const __bf16* Vtb = Vt + (size_t)bh * DKH * SS;

  f32x8 acc[4] = {};
  for (int k0 = 0; k0 < SS; k0 += 32) {
    bf16x16 afrag;
    const __bf16* prow = Pb + (size_t)(q0 + l15) * SS + k0;
    afrag = *(const bf16x16*)&prow[half * 16];
#pragma unroll
    for (int t = 0; t < 4; ++t) {
      int d = t * 16 + l15;
      bf16x16 bfrag = *(const bf16x16*)&Vtb[(size_t)d * SS + k0 + half * 16];
      acc[t] = __builtin_amdgcn_wmma_f32_16x16x32_bf16(
          false, afrag, false, bfrag, (short)0, acc[t], false, false);
    }
  }
#pragma unroll
  for (int t = 0; t < 4; ++t) {
    int d = t * 16 + l15;
#pragma unroll
    for (int v = 0; v < 8; ++v) {
      int q = q0 + half * 8 + v;
      O[((size_t)(b * SS + q)) * DIM + h * DKH + d] = (__bf16)acc[t][v];
    }
  }
}

// ---------------------------------------------------------------------------
// Fused residual add + LayerNorm, fp32 math; emits fp32 state + bf16 copy.
// One D=512 row per 256-thread block (2 elems/thread).
// ---------------------------------------------------------------------------
__launch_bounds__(256)
__global__ void add_ln_kernel(const float* __restrict__ x,
                              const float* __restrict__ delta,
                              const float* __restrict__ g,
                              const float* __restrict__ bta,
                              float* __restrict__ outf,
                              __bf16* __restrict__ outb) {
  __shared__ float red[256];
  __shared__ float s_mean, s_rstd;
  size_t row = blockIdx.x;
  int t = threadIdx.x;
  const float* xp = x + row * DIM;
  const float* dp = delta + row * DIM;
  float v0 = xp[t] + dp[t];
  float v1 = xp[t + 256] + dp[t + 256];
  red[t] = v0 + v1;
  __syncthreads();
  for (int s = 128; s > 0; s >>= 1) {
    if (t < s) red[t] += red[t + s];
    __syncthreads();
  }
  if (t == 0) s_mean = red[0] * (1.0f / DIM);
  __syncthreads();
  float m = s_mean;
  float d0 = v0 - m, d1 = v1 - m;
  red[t] = d0 * d0 + d1 * d1;
  __syncthreads();
  for (int s = 128; s > 0; s >>= 1) {
    if (t < s) red[t] += red[t + s];
    __syncthreads();
  }
  if (t == 0) s_rstd = rsqrtf(red[0] * (1.0f / DIM) + EPSV);
  __syncthreads();
  float rs = s_rstd;
  float o0 = d0 * rs * g[t] + bta[t];
  float o1 = d1 * rs * g[t + 256] + bta[t + 256];
  outf[row * DIM + t] = o0;
  outf[row * DIM + t + 256] = o1;
  outb[row * DIM + t] = (__bf16)o0;
  outb[row * DIM + t + 256] = (__bf16)o1;
}

// ---------------------------------------------------------------------------
// Host-side orchestration
// ---------------------------------------------------------------------------
extern "C" void kernel_launch(void* const* d_in, const int* in_sizes, int n_in,
                              void* d_out, int out_size, void* d_ws,
                              size_t ws_size, hipStream_t stream) {
  const float* src   = (const float*)d_in[0];
  const float* Wq    = (const float*)d_in[1];
  const float* bq    = (const float*)d_in[2];
  const float* Wk    = (const float*)d_in[3];
  const float* bk    = (const float*)d_in[4];
  const float* Wv    = (const float*)d_in[5];
  const float* bv    = (const float*)d_in[6];
  const float* Wo    = (const float*)d_in[7];
  const float* bo    = (const float*)d_in[8];
  const float* ln1g  = (const float*)d_in[9];
  const float* ln1b  = (const float*)d_in[10];
  const float* ln2g  = (const float*)d_in[11];
  const float* ln2b  = (const float*)d_in[12];
  const float* Wf1   = (const float*)d_in[13];
  const float* bf1   = (const float*)d_in[14];
  const float* Wf2   = (const float*)d_in[15];
  const float* bf2   = (const float*)d_in[16];
  float* outp = (float*)d_out;

  // Deterministic workspace carving (256B aligned)
  char* w = (char*)d_ws;
  auto carve = [&](size_t bytes) -> void* {
    void* p = (void*)w;
    w += (bytes + 255) & ~(size_t)255;
    return p;
  };
  const size_t tokd  = (size_t)MTOK * DIM;
  const size_t attnN = (size_t)BH * SS * SS;

  float*  scores = (float*)carve(attnN * 4);          // residual pre-softmax
  __bf16* probs  = (__bf16*)carve(attnN * 2);
  float*  xf     = (float*)carve(tokd * 4);           // fp32 state
  float*  tmpf   = (float*)carve(tokd * 4);           // proj / ffn2 output
  __bf16* xb     = (__bf16*)carve(tokd * 2);          // bf16 state
  __bf16* Qb     = (__bf16*)carve(tokd * 2);
  __bf16* Kb     = (__bf16*)carve(tokd * 2);
  __bf16* Vb     = (__bf16*)carve(tokd * 2);
  __bf16* Vt     = (__bf16*)carve(tokd * 2);          // [bh, d, kk]
  __bf16* Ob     = (__bf16*)carve(tokd * 2);
  __bf16* hb     = (__bf16*)carve((size_t)MTOK * FFD * 2);
  __bf16* wqt    = (__bf16*)carve((size_t)LYR * DIM * DIM * 2);   // transposed
  __bf16* wkt    = (__bf16*)carve((size_t)LYR * DIM * DIM * 2);
  __bf16* wvt    = (__bf16*)carve((size_t)LYR * DIM * DIM * 2);
  __bf16* wot    = (__bf16*)carve((size_t)LYR * DIM * DIM * 2);
  __bf16* wf1t   = (__bf16*)carve((size_t)LYR * DIM * FFD * 2);
  __bf16* wf2t   = (__bf16*)carve((size_t)LYR * FFD * DIM * 2);
  (void)ws_size; (void)in_sizes; (void)n_in; (void)out_size;

  // Weight convert + transpose: [K,N] f32 -> [N,K] bf16, batched over layers
  dim3 gWdd(DIM / 32, DIM / 32, LYR);
  transpose_cvt_kernel<<<gWdd, 256, 0, stream>>>(Wq,  wqt,  DIM, DIM);
  transpose_cvt_kernel<<<gWdd, 256, 0, stream>>>(Wk,  wkt,  DIM, DIM);
  transpose_cvt_kernel<<<gWdd, 256, 0, stream>>>(Wv,  wvt,  DIM, DIM);
  transpose_cvt_kernel<<<gWdd, 256, 0, stream>>>(Wo,  wot,  DIM, DIM);
  dim3 gWf1(FFD / 32, DIM / 32, LYR);
  transpose_cvt_kernel<<<gWf1, 256, 0, stream>>>(Wf1, wf1t, DIM, FFD);
  dim3 gWf2(DIM / 32, FFD / 32, LYR);
  transpose_cvt_kernel<<<gWf2, 256, 0, stream>>>(Wf2, wf2t, FFD, DIM);
  f32_to_bf16_kernel<<<2048, 256, 0, stream>>>(src, xb, (int)tokd);

  const float scale = 1.0f / sqrtf((float)DKH);
  dim3 gProj(DIM / 128, MTOK / 64);   // N=512 GEMMs
  dim3 gFF1(FFD / 128, MTOK / 64);    // N=2048 GEMM
  dim3 gScores(SS / 16, SS / 128, BH);
  dim3 gVt(SS / 32, DKH / 32, BH);
  dim3 gAttnO(SS / 128, BH);

  for (int i = 0; i < LYR; ++i) {
    const size_t wdd = (size_t)i * DIM * DIM;
    const size_t wdf = (size_t)i * DIM * FFD;

    // QKV projections (bf16 WMMA, fp32 accumulate, bf16 out)
    gemm_bf16_kernel<<<gProj, 256, 0, stream>>>(xb, wqt + wdd, bq + i * DIM,
                                                nullptr, Qb, MTOK, DIM, DIM, 0);
    gemm_bf16_kernel<<<gProj, 256, 0, stream>>>(xb, wkt + wdd, bk + i * DIM,
                                                nullptr, Kb, MTOK, DIM, DIM, 0);
    gemm_bf16_kernel<<<gProj, 256, 0, stream>>>(xb, wvt + wdd, bv + i * DIM,
                                                nullptr, Vb, MTOK, DIM, DIM, 0);

    // scores (+)= scale * Q K^T   (residual across layers)
    attn_scores_kernel<<<gScores, 256, 0, stream>>>(Qb, Kb, scores,
                                                    (i == 0) ? 1 : 0, scale);
    // softmax -> bf16 probabilities
    softmax_kernel<<<(unsigned)(BH * SS), 256, 0, stream>>>(scores, probs);
    // V -> Vt (K-contiguous for P*V), then O = P V
    transpose_v_kernel<<<gVt, 256, 0, stream>>>(Vb, Vt);
    attn_out_kernel<<<gAttnO, 256, 0, stream>>>(probs, Vt, Ob);

    // output projection -> fp32
    gemm_bf16_kernel<<<gProj, 256, 0, stream>>>(Ob, wot + wdd, bo + i * DIM,
                                                tmpf, nullptr, MTOK, DIM, DIM, 0);
    // x = LN(x + attn_out)
    const float* xin = (i == 0) ? src : xf;
    add_ln_kernel<<<MTOK, 256, 0, stream>>>(xin, tmpf, ln1g + i * DIM,
                                            ln1b + i * DIM, xf, xb);

    // FFN: h = GELU(x Wf1 + b1) in bf16, then ff = h Wf2 + b2 in fp32
    gemm_bf16_kernel<<<gFF1, 256, 0, stream>>>(xb, wf1t + wdf, bf1 + i * FFD,
                                               nullptr, hb, MTOK, FFD, DIM,
                                               FLAG_GELU);
    gemm_bf16_kernel<<<gProj, 256, 0, stream>>>(hb, wf2t + wdf, bf2 + i * DIM,
                                                tmpf, nullptr, MTOK, DIM, FFD, 0);
    // x = LN(x + ff); last layer writes final fp32 output directly
    float* lnout = (i == LYR - 1) ? outp : xf;
    add_ln_kernel<<<MTOK, 256, 0, stream>>>(xf, tmpf, ln2g + i * DIM,
                                            ln2b + i * DIM, lnout, xb);
  }
}
